// MultiheadAttention_62654982914587
// MI455X (gfx1250) — matrix operand order repaired and checked by reading, not compiled, not run on hardware
//
#include <hip/hip_runtime.h>
#include <hip/hip_bf16.h>
#include <type_traits>

// ---------------------------------------------------------------------------
// CDNA5 (gfx1250) multi-head attention, bf16 WMMA everywhere.
//   - weights pre-converted to bf16 once; GEMM B-tiles staged with
//     global_load_async_to_lds_b128 (ASYNCcnt), zero conversion in hot loop
//   - flash attention: double-buffered async K tiles, V pipelined via regs
// ---------------------------------------------------------------------------

typedef __bf16 bf16;
typedef __attribute__((ext_vector_type(16))) __bf16 bf16x16;
typedef __attribute__((ext_vector_type(8)))  __bf16 bf16x8;
typedef __attribute__((ext_vector_type(4)))  __bf16 bf16x4;
typedef __attribute__((ext_vector_type(8)))  float  floatx8;

// ---- WMMA wrapper ---------------------------------------------------------
__device__ __forceinline__ floatx8 wmma_bf16(bf16x16 a, bf16x16 b, floatx8 c) {
  return __builtin_amdgcn_wmma_f32_16x16x32_bf16(false, a, false, b,
                                                 (short)0, c, false, false);
}

__device__ __forceinline__ bf16x16 cat8(bf16x8 lo, bf16x8 hi) {
  return __builtin_shufflevector(lo, hi, 0, 1, 2, 3, 4, 5, 6, 7,
                                 8, 9, 10, 11, 12, 13, 14, 15);
}

// A-matrix 16x32 bf16 fragment from an m-major [16][ld] region.
__device__ __forceinline__ bf16x16 frag_a(const bf16* p, int ld, int lane) {
  const int m = lane & 15, half = lane >> 4;
  const bf16x8 lo = *(const bf16x8*)&p[m * ld + 8 * half];
  const bf16x8 hi = *(const bf16x8*)&p[m * ld + 16 + 8 * half];
  return cat8(lo, hi);
}

// B-matrix 32x16 bf16 fragment, source stored n-major: elem(k,n)=p[n*ld+k].
__device__ __forceinline__ bf16x16 frag_b_nmajor(const bf16* p, int ld, int lane) {
  const int n = lane & 15, half = lane >> 4;
  const bf16x8 lo = *(const bf16x8*)&p[n * ld + 8 * half];
  const bf16x8 hi = *(const bf16x8*)&p[n * ld + 16 + 8 * half];
  return cat8(lo, hi);
}

// 16B async global->LDS copy (ASYNCcnt-tracked), one lane-chunk.
__device__ __forceinline__ void async_b128(const bf16* g, bf16* l) {
  const unsigned lds_off = (unsigned)(uintptr_t)l;
  const unsigned long long ga = (unsigned long long)(uintptr_t)g;
  asm volatile("global_load_async_to_lds_b128 %0, %1, off"
               :: "v"(lds_off), "v"(ga) : "memory");
}

// ---------------------------------------------------------------------------
// Elementwise f32 -> bf16 (weight pre-conversion).
// ---------------------------------------------------------------------------
__global__ __launch_bounds__(256) void cvt_f32_bf16(
    const float* __restrict__ s, bf16* __restrict__ d, int n) {
  const int i = (blockIdx.x * 256 + threadIdx.x) * 4;
  if (i < n) {
    const float4 f = *(const float4*)&s[i];
    bf16x4 o;
    o[0] = (bf16)f.x; o[1] = (bf16)f.y; o[2] = (bf16)f.z; o[3] = (bf16)f.w;
    *(bf16x4*)&d[i] = o;
  }
}

// ---------------------------------------------------------------------------
// GEMM: Y[M,N] = X[M,K] @ Wb[N,K]^T + bias[N]   (torch Linear)
// BM=128 BN=128 BK=64, 256 threads = 8 waves, wave tile 32x64.
// Wb is bf16 (pre-converted) -> B staging is async DMA.
// ---------------------------------------------------------------------------
template <typename InT, typename OutT>
__global__ __launch_bounds__(256) void wmma_gemm_xwt_bias(
    const InT* __restrict__ X, const bf16* __restrict__ Wb,
    const float* __restrict__ bias, OutT* __restrict__ Y,
    int M, int N, int K) {
  constexpr int BM = 128, BN = 128, BK = 64;
  constexpr int LDA = 72, LDB = 72;  // bf16 units; 144B pitch (16B aligned)
  __shared__ __align__(16) bf16 sA[BM * LDA];
  __shared__ __align__(16) bf16 sB[BN * LDB];

  const int tid  = threadIdx.x;
  const int lane = tid & 31;
  const int wave = tid >> 5;   // 0..7
  const int wm   = wave >> 1;  // 0..3 : 32-row group
  const int wn   = wave & 1;   // 0..1 : 64-col group

  const int m0 = blockIdx.y * BM;
  const int n0 = blockIdx.x * BN;

  floatx8 acc[2][4] = {};

  for (int k0 = 0; k0 < K; k0 += BK) {
    // ---- B tile (bf16 weights): async global->LDS DMA, 16B chunks ----
#pragma unroll
    for (int j = 0; j < 4; ++j) {
      const int i = tid + j * 256;                // 1024 chunks of 8 bf16
      const int r = i >> 3, c8 = (i & 7) << 3;
      async_b128(&Wb[(size_t)(n0 + r) * K + k0 + c8], &sB[r * LDB + c8]);
    }
    // ---- A tile ----
    if constexpr (std::is_same<InT, bf16>::value) {
      // already bf16: async DMA as well
#pragma unroll
      for (int j = 0; j < 4; ++j) {
        const int i = tid + j * 256;
        const int r = i >> 3, c8 = (i & 7) << 3;
        async_b128(&X[(size_t)(m0 + r) * K + k0 + c8], &sA[r * LDA + c8]);
      }
    } else {
      // f32 source: convert while staging
      for (int i = tid; i < BM * (BK / 4); i += 256) {
        const int r = i >> 4, c = (i & 15) << 2;
        const InT* src = &X[(size_t)(m0 + r) * K + k0 + c];
        bf16* dst = &sA[r * LDA + c];
#pragma unroll
        for (int u = 0; u < 4; ++u) dst[u] = (bf16)(float)src[u];
      }
      if (k0 + BK < K)
        __builtin_prefetch(&X[(size_t)(m0 + (tid >> 1)) * K + k0 + BK], 0, 1);
    }
    asm volatile("s_wait_asynccnt 0x0" ::: "memory");
    __syncthreads();

#pragma unroll
    for (int kc = 0; kc < 2; ++kc) {
      bf16x16 bf[4];
#pragma unroll
      for (int j = 0; j < 4; ++j)
        bf[j] = frag_b_nmajor(&sB[(wn * 64 + j * 16) * LDB + kc * 32], LDB, lane);
#pragma unroll
      for (int i = 0; i < 2; ++i) {
        const bf16x16 a = frag_a(&sA[(wm * 32 + i * 16) * LDA + kc * 32], LDA, lane);
#pragma unroll
        for (int j = 0; j < 4; ++j) acc[i][j] = wmma_bf16(a, bf[j], acc[i][j]);
      }
    }
    __syncthreads();
  }

  // Epilogue: C layout = lane holds col n=lane&15; VGPR r -> row r+8*half.
  const int half = lane >> 4, nn = lane & 15;
#pragma unroll
  for (int i = 0; i < 2; ++i) {
#pragma unroll
    for (int j = 0; j < 4; ++j) {
      const int ncol = n0 + wn * 64 + j * 16 + nn;
      const float bv = bias[ncol];
#pragma unroll
      for (int r = 0; r < 8; ++r) {
        const int mrow = m0 + wm * 32 + i * 16 + r + 8 * half;
        Y[(size_t)mrow * N + ncol] = (OutT)(acc[i][j][r] + bv);
      }
    }
  }
}

// ---------------------------------------------------------------------------
// Flash attention helpers (128-thread block).
// ---------------------------------------------------------------------------
__device__ __forceinline__ void issue_k_async(const bf16* Kb, size_t rowBase,
                                              int D, int colBase, bf16* dstK,
                                              int ld, int tid) {
#pragma unroll
  for (int j = 0; j < 4; ++j) {
    const int c = tid + j * 128;                // 512 chunks of 8 bf16
    const int r = c >> 3, c8 = (c & 7) << 3;
    async_b128(&Kb[(rowBase + r) * D + colBase + c8], &dstK[r * ld + c8]);
  }
}

__device__ __forceinline__ void load_v_regs(const bf16* Vb, size_t rowBase,
                                            int D, int colBase, int tid,
                                            bf16x4 v[4]) {
#pragma unroll
  for (int j = 0; j < 4; ++j) {
    const int i = tid + j * 128;
    const int r = i >> 4, c4 = (i & 15) << 2;
    v[j] = *(const bf16x4*)&Vb[(rowBase + r) * D + colBase + c4];
  }
}

__device__ __forceinline__ void store_v_trans(bf16* sVt, int ld, int tid,
                                              const bf16x4 v[4]) {
#pragma unroll
  for (int j = 0; j < 4; ++j) {
    const int i = tid + j * 128;
    const int r = i >> 4, c4 = (i & 15) << 2;
#pragma unroll
    for (int u = 0; u < 4; ++u) sVt[(c4 + u) * ld + r] = v[j][u];
  }
}

// ---------------------------------------------------------------------------
// Flash attention: block = (b, h, 64 q-rows); 4 waves, 16 q-rows each. hd=64.
// Double-buffered async K tiles; V global loads pipelined through registers.
// ---------------------------------------------------------------------------
__global__ __launch_bounds__(128) void wmma_flash_attn(
    const bf16* __restrict__ Qb, const bf16* __restrict__ Kb,
    const bf16* __restrict__ Vb, bf16* __restrict__ Ob, int S, int H) {
  constexpr int LDK = 72, LDV = 72, LDP = 72;
  __shared__ __align__(16) bf16 sK0[64 * LDK];    // [key][hd], buffer 0
  __shared__ __align__(16) bf16 sK1[64 * LDK];    // [key][hd], buffer 1
  __shared__ __align__(16) bf16 sVt[64 * LDV];    // [hd][key] (transposed)
  __shared__ __align__(16) bf16 sP[4][16 * LDP];  // per-wave [qrow][key]

  const int tid = threadIdx.x, lane = tid & 31, wave = tid >> 5;
  const int h = blockIdx.y, b = blockIdx.z;
  const int D = H * 64;
  const size_t baseRow = (size_t)b * S;
  const int colBase = h * 64;
  const int q0 = blockIdx.x * 64 + wave * 16;

  const int m = lane & 15, half = lane >> 4, nn = lane & 15;

  // Resident Q fragments (A operand), 2 K-chunks covering hd=64.
  bf16x16 qf[2];
#pragma unroll
  for (int kc = 0; kc < 2; ++kc) {
    const bf16* qrow = &Qb[(baseRow + q0 + m) * D + colBase + kc * 32];
    qf[kc] = cat8(*(const bf16x8*)&qrow[8 * half],
                  *(const bf16x8*)&qrow[16 + 8 * half]);
  }

  floatx8 o[4] = {};
  float mrow[8], lrow[8];
#pragma unroll
  for (int r = 0; r < 8; ++r) { mrow[r] = -3.0e38f; lrow[r] = 0.0f; }

  const float scale = 0.125f;  // 1/sqrt(64)
  const int NT = S / 64;

  // ---- Prologue: start tile 0 ----
  bf16x4 vreg[4], vnext[4];
  issue_k_async(Kb, baseRow, D, colBase, sK0, LDK, tid);
  load_v_regs(Vb, baseRow, D, colBase, tid, vreg);

  for (int it = 0; it < NT; ++it) {
    bf16* curK = (it & 1) ? sK1 : sK0;
    bf16* nxtK = (it & 1) ? sK0 : sK1;

    // Commit this tile's V (regs -> transposed LDS).
    store_v_trans(sVt, LDV, tid, vreg);

    // Kick off tile it+1 while we compute tile it.
    if (it + 1 < NT) {
      const size_t nrow = baseRow + (size_t)(it + 1) * 64;
      issue_k_async(Kb, nrow, D, colBase, nxtK, LDK, tid);
      load_v_regs(Vb, nrow, D, colBase, tid, vnext);
      // async loads retire in order: <=4 outstanding => tile `it` landed.
      asm volatile("s_wait_asynccnt 0x4" ::: "memory");
    } else {
      asm volatile("s_wait_asynccnt 0x0" ::: "memory");
    }
    __syncthreads();

    // ---- Scores: S = (Q @ K^T) * scale ; 4 key sub-tiles of 16 ----
    floatx8 s[4];
#pragma unroll
    for (int t = 0; t < 4; ++t) {
      floatx8 c = {};
#pragma unroll
      for (int kc = 0; kc < 2; ++kc) {
        const bf16x16 bk =
            frag_b_nmajor(&curK[(t * 16) * LDK + kc * 32], LDK, lane);
        c = wmma_bf16(qf[kc], bk, c);
      }
      s[t] = c;
    }

    // ---- Online softmax (row = r + 8*half, cols across 16 lanes) ----
#pragma unroll
    for (int t = 0; t < 4; ++t)
#pragma unroll
      for (int r = 0; r < 8; ++r) s[t][r] *= scale;

#pragma unroll
    for (int r = 0; r < 8; ++r) {
      float v = fmaxf(fmaxf(s[0][r], s[1][r]), fmaxf(s[2][r], s[3][r]));
#pragma unroll
      for (int off = 8; off >= 1; off >>= 1)
        v = fmaxf(v, __shfl_xor(v, off, 16));
      const float nm = fmaxf(mrow[r], v);
      const float alpha = __expf(mrow[r] - nm);
      mrow[r] = nm;
      float rs = 0.0f;
#pragma unroll
      for (int t = 0; t < 4; ++t) {
        const float p = __expf(s[t][r] - nm);
        s[t][r] = p;
        rs += p;
      }
#pragma unroll
      for (int off = 8; off >= 1; off >>= 1) rs += __shfl_xor(rs, off, 16);
      lrow[r] = lrow[r] * alpha + rs;
#pragma unroll
      for (int nt = 0; nt < 4; ++nt) o[nt][r] *= alpha;
    }

    // ---- Reshape P (C layout) -> A layout via per-wave LDS scratch ----
#pragma unroll
    for (int t = 0; t < 4; ++t)
#pragma unroll
      for (int r = 0; r < 8; ++r)
        sP[wave][(r + 8 * half) * LDP + t * 16 + nn] = (bf16)s[t][r];
    __syncthreads();

    // ---- O += P @ V ----
#pragma unroll
    for (int kc = 0; kc < 2; ++kc) {
      const bf16x16 pa = frag_a(&sP[wave][kc * 32], LDP, lane);
#pragma unroll
      for (int nt = 0; nt < 4; ++nt) {
        const bf16x16 vbf =
            frag_b_nmajor(&sVt[(nt * 16) * LDV + kc * 32], LDV, lane);
        o[nt] = wmma_bf16(pa, vbf, o[nt]);
      }
    }
    __syncthreads();

    if (it + 1 < NT) {
#pragma unroll
      for (int j = 0; j < 4; ++j) vreg[j] = vnext[j];
    }
  }

  // ---- Normalize and store bf16 attention output [b, q, h, hd] ----
#pragma unroll
  for (int r = 0; r < 8; ++r) {
    const float inv = 1.0f / lrow[r];
    const int qrow = q0 + r + 8 * half;
#pragma unroll
    for (int nt = 0; nt < 4; ++nt)
      Ob[(baseRow + qrow) * D + colBase + nt * 16 + nn] =
          (bf16)(o[nt][r] * inv);
  }
}

// ---------------------------------------------------------------------------
// Launch: weight converts -> 3 projection GEMMs -> attention -> output GEMM.
// ---------------------------------------------------------------------------
extern "C" void kernel_launch(void* const* d_in, const int* in_sizes, int n_in,
                              void* d_out, int out_size, void* d_ws,
                              size_t ws_size, hipStream_t stream) {
  (void)in_sizes; (void)n_in; (void)out_size; (void)ws_size;
  const float* Q  = (const float*)d_in[0];
  const float* K  = (const float*)d_in[1];
  const float* V  = (const float*)d_in[2];
  const float* Wq = (const float*)d_in[3];
  const float* bq = (const float*)d_in[4];
  const float* Wk = (const float*)d_in[5];
  const float* bk = (const float*)d_in[6];
  const float* Wv = (const float*)d_in[7];
  const float* bv = (const float*)d_in[8];
  const float* Wo = (const float*)d_in[9];
  const float* bo = (const float*)d_in[10];
  float* out = (float*)d_out;

  const int B = 2, S = 2048, D = 1024, H = 16;
  const int M = B * S;
  const size_t elems = (size_t)M * D;   // 4M
  const size_t welem = (size_t)D * D;   // 1M

  bf16* qb  = (bf16*)d_ws;
  bf16* kb  = qb + elems;
  bf16* vb  = kb + elems;
  bf16* ab  = vb + elems;   // attention output, bf16
  bf16* wqb = ab + elems;
  bf16* wkb = wqb + welem;
  bf16* wvb = wkb + welem;
  bf16* wob = wvb + welem;

  // Pre-convert weights to bf16 (one-time, bandwidth-trivial).
  dim3 cblk(256), cgrid((unsigned)(welem / 4 / 256));
  cvt_f32_bf16<<<cgrid, cblk, 0, stream>>>(Wq, wqb, (int)welem);
  cvt_f32_bf16<<<cgrid, cblk, 0, stream>>>(Wk, wkb, (int)welem);
  cvt_f32_bf16<<<cgrid, cblk, 0, stream>>>(Wv, wvb, (int)welem);
  cvt_f32_bf16<<<cgrid, cblk, 0, stream>>>(Wo, wob, (int)welem);

  dim3 gblk(256), ggrid(D / 128, M / 128);
  wmma_gemm_xwt_bias<float, bf16><<<ggrid, gblk, 0, stream>>>(Q, wqb, bq, qb, M, D, D);
  wmma_gemm_xwt_bias<float, bf16><<<ggrid, gblk, 0, stream>>>(K, wkb, bk, kb, M, D, D);
  wmma_gemm_xwt_bias<float, bf16><<<ggrid, gblk, 0, stream>>>(V, wvb, bv, vb, M, D, D);

  dim3 ablk(128), agrid(S / 64, H, B);
  wmma_flash_attn<<<agrid, ablk, 0, stream>>>(qb, kb, vb, ab, S, H);

  wmma_gemm_xwt_bias<bf16, float><<<ggrid, gblk, 0, stream>>>(ab, wob, bo, out, M, D, D);
}
